// SSMEncoder_87978110091747
// MI455X (gfx1250) — matrix operand order repaired and checked by reading, not compiled, not run on hardware
//
#include <hip/hip_runtime.h>
#include <hip/hip_bf16.h>
#include <math.h>

// ---------------- problem constants ----------------
#define BB       8
#define CIN      3
#define DM       512
#define N2       32
#define NLAY     4
#define LATENT   256
#define LL       4096
#define EPSLN    1e-5f

typedef __attribute__((ext_vector_type(16))) _Float16 v16h;
typedef __attribute__((ext_vector_type(8)))  float    v8f;

union AFrag { v16h v; _Float16 h[16]; uint4 q[2]; };

// ---------------------------------------------------------------------------
// Kernel 0: fp32 -> f16 bulk convert (used for Wout and Wstats)
// ---------------------------------------------------------------------------
__global__ void k_cvt(const float* __restrict__ src, _Float16* __restrict__ dst, int n) {
    int i = blockIdx.x * blockDim.x + threadIdx.x;
    if (i < n) dst[i] = (_Float16)src[i];
}

// Zero rows 8..15 of the padded f16 mean matrix (16 x 512)
__global__ void k_padzero(_Float16* __restrict__ hmH) {
    int i = blockIdx.x * blockDim.x + threadIdx.x;
    if (i < 8 * DM) hmH[8 * DM + i] = (_Float16)0.0f;
}

// ---------------------------------------------------------------------------
// Kernel 1: input 1x1 conv projection + pos_emb -> h (B, D, L) fp32
// ---------------------------------------------------------------------------
__global__ void k_proj(const float* __restrict__ x, const float* __restrict__ Wp,
                       const float* __restrict__ bp, const float* __restrict__ pos,
                       float* __restrict__ h) {
    int idx = blockIdx.x * blockDim.x + threadIdx.x;          // (b,d,l) flat
    if (idx >= BB * DM * LL) return;
    int l = idx & (LL - 1);
    int d = (idx >> 12) & (DM - 1);
    int b = idx >> 21;
    const float* xb = x + (size_t)b * CIN * LL + l;
    float v = fmaf(Wp[d * 3 + 0], xb[0],
              fmaf(Wp[d * 3 + 1], xb[LL],
              fmaf(Wp[d * 3 + 2], xb[2 * LL], bp[d])));
    h[idx] = v + pos[(size_t)l * DM + d];
}

// ---------------------------------------------------------------------------
// Kernel 2: S4D recurrence + Dskip + exact GELU -> yactT f16 (B, L, D)
// Block = 256 threads = 8 waves; wave w handles channel (b, d0+w), one lane
// per complex mode. Output is written TRANSPOSED (D-contiguous) via LDS.
// ---------------------------------------------------------------------------
__global__ void k_ssm_gelu(const float* __restrict__ h,
                           const float* __restrict__ log_dt,
                           const float* __restrict__ log_A_real,
                           const float* __restrict__ A_imag,
                           const float* __restrict__ C_re,
                           const float* __restrict__ C_im,
                           const float* __restrict__ Dskip,
                           _Float16* __restrict__ yactT, int layer) {
    __shared__ _Float16 sT[8][33];            // 8 channels x 32 timesteps (+pad)

    const int w    = threadIdx.x >> 5;        // wave = channel offset 0..7
    const int lane = threadIdx.x & 31;        // mode
    const int b  = blockIdx.x >> 6;           // 64 blocks per batch
    const int d0 = (blockIdx.x & 63) * 8;
    const int d  = d0 + w;

    int pc = layer * DM + d;
    int pm = pc * N2 + lane;

    float dt  = __expf(log_dt[pc]);
    float Ar  = -__expf(log_A_real[pm]);
    float Ai  = A_imag[pm];
    float er  = __expf(dt * Ar);
    float lr  = er * __cosf(dt * Ai);
    float li  = er * __sinf(dt * Ai);
    float wr = lr - 1.0f, wi = li;
    float inv = 1.0f / (Ar * Ar + Ai * Ai);
    float qr = (wr * Ar + wi * Ai) * inv;
    float qi = (wi * Ar - wr * Ai) * inv;
    float cr = C_re[pm], ci = C_im[pm];
    float C2r = 2.0f * (cr * qr - ci * qi);
    float C2i = 2.0f * (cr * qi + ci * qr);
    float Dk  = Dskip[pc];

    const float* u = h + ((size_t)b * DM + d) * LL;

    float sr = 0.0f, si = 0.0f;
    for (int t0 = 0; t0 < LL; t0 += 32) {
        float uv = u[t0 + lane];              // coalesced chunk load
        float yreg = 0.0f;
#pragma unroll
        for (int tau = 0; tau < 32; ++tau) {
            float ub = __shfl(uv, tau, 32);
            float nsr = fmaf(lr, sr, fmaf(-li, si, ub));
            float nsi = fmaf(lr, si, li * sr);
            sr = nsr; si = nsi;
            float p = fmaf(C2r, sr, -C2i * si);
            p += __shfl_xor(p, 1, 32);
            p += __shfl_xor(p, 2, 32);
            p += __shfl_xor(p, 4, 32);
            p += __shfl_xor(p, 8, 32);
            p += __shfl_xor(p, 16, 32);
            if (lane == tau) yreg = fmaf(Dk, ub, p);
        }
        float g = 0.5f * yreg * (1.0f + erff(yreg * 0.70710678118654752f));
        sT[w][lane] = (_Float16)g;            // lane tau holds y[t0+tau]
        __syncthreads();
        {   // transposed store: (t = i>>3, dd = i&7) -> yactT[b, t0+t, d0+dd]
            int t  = threadIdx.x >> 3;
            int dd = threadIdx.x & 7;
            yactT[((size_t)b * LL + t0 + t) * DM + d0 + dd] = sT[dd][t];
        }
        __syncthreads();
    }
}

// ---------------------------------------------------------------------------
// Kernel 3: WMMA GEMM y2 = Wout(1024x512) x yact(512xL) + bout, fused GLU:
//   z = y2[0:512] * sigmoid(y2[512:1024]);  h += z   (residual)
// ---------------------------------------------------------------------------
__global__ void k_gemm_glu(const _Float16* __restrict__ WoutH,
                           const _Float16* __restrict__ yactT,
                           const float* __restrict__ bout,
                           float* __restrict__ h, int layer) {
    __shared__ float sA[64][64];              // 'a' results
    __shared__ float sG[64][64];              // gate results

    const int l0 = blockIdx.x * 64;
    const int d0 = blockIdx.y * 64;
    const int b  = blockIdx.z;
    const int w    = threadIdx.x >> 5;
    const int lane = threadIdx.x & 31;

    const int obase = (w < 4) ? (d0 + 16 * w) : (DM + d0 + 16 * (w - 4));
    const _Float16* Wrow = WoutH + ((size_t)(layer * 2 * DM + obase) + (lane & 15)) * DM;
    const int kbase = (lane < 16) ? 0 : 16;   // A-operand contiguous k layout
    const int koffB = (lane < 16) ? 0 : 16;   // B-operand contiguous k layout (same as A)
    const _Float16* Bbase = yactT + ((size_t)b * LL + l0 + (lane & 15)) * DM + koffB;

    v8f acc0 = {}, acc1 = {}, acc2 = {}, acc3 = {};

#pragma unroll 4
    for (int kk = 0; kk < DM; kk += 32) {
        AFrag a;
        a.q[0] = *(const uint4*)(Wrow + kk + kbase);
        a.q[1] = *(const uint4*)(Wrow + kk + 8 + kbase);
        if (kk + 32 < DM) __builtin_prefetch(Wrow + kk + 32, 0, 3);

#pragma unroll
        for (int t = 0; t < 4; ++t) {
            const _Float16* bp = Bbase + (size_t)(16 * t) * DM + kk;
            AFrag bf;
            bf.q[0] = *(const uint4*)(bp);
            bf.q[1] = *(const uint4*)(bp + 8);
            if (kk + 32 < DM) __builtin_prefetch(bp + 32, 0, 3);
            v8f* accp = (t == 0) ? &acc0 : (t == 1) ? &acc1 : (t == 2) ? &acc2 : &acc3;
            *accp = __builtin_amdgcn_wmma_f32_16x16x32_f16(
                false, a.v, false, bf.v, (short)0, *accp, false, false);
        }
    }

    // ---- bias + stash to LDS for GLU pairing ----
    float (*dst)[64] = (w < 4) ? sA : sG;
    const int mhi = (lane >> 4) * 8;
    const int lrb = (w & 3) * 16;
    const int coln = lane & 15;
#pragma unroll
    for (int t = 0; t < 4; ++t) {
        v8f acc = (t == 0) ? acc0 : (t == 1) ? acc1 : (t == 2) ? acc2 : acc3;
        int col = 16 * t + coln;
#pragma unroll
        for (int r = 0; r < 8; ++r) {
            int m = r + mhi;
            dst[lrb + m][col] = acc[r] + bout[layer * 2 * DM + obase + m];
        }
    }
    __syncthreads();

    // ---- GLU + residual ----
#pragma unroll
    for (int j = 0; j < 16; ++j) {
        int idx = threadIdx.x + 256 * j;
        int lr = idx >> 6, col = idx & 63;
        float av = sA[lr][col];
        float gv = sG[lr][col];
        float z  = av * (1.0f / (1.0f + __expf(-gv)));
        size_t hoff = ((size_t)b * DM + d0 + lr) * LL + l0 + col;
        h[hoff] += z;
    }
}

// ---------------------------------------------------------------------------
// Kernel 4: LayerNorm over D, LDS-tiled for coalescing.
// One workgroup per (b, 16-wide l-slab). The 512x16 f32 tile is staged
// memory->LDS with GLOBAL_LOAD_ASYNC_TO_LDS_B128 (ASYNCcnt), reduced over D
// out of LDS, and written back LDS->memory with GLOBAL_STORE_ASYNC_FROM_LDS.
// LDS pitch = 20 floats (80B) keeps every b128 16B-aligned.
// ---------------------------------------------------------------------------
#define LNP 20
__global__ void k_layernorm(float* __restrict__ h, const float* __restrict__ g,
                            const float* __restrict__ bta, int layer) {
    __shared__ float sT[DM * LNP];            // 40 KB

    const int b  = blockIdx.x >> 8;           // 256 l-slabs per batch
    const int l0 = (blockIdx.x & 255) * 16;
    const int w    = threadIdx.x >> 5;
    const int lane = threadIdx.x & 31;
    const unsigned sbase = (unsigned)(size_t)(&sT[0]);   // LDS byte offset (addr[31:0])

    // ---- stage tile: 2048 float4, coalesced 64B rows -> LDS (async) ----
#pragma unroll
    for (int j = 0; j < 8; ++j) {
        int e  = threadIdx.x + 256 * j;       // 0..2047
        int d  = e >> 2;                      // row
        int lq = e & 3;                       // float4 within row
        unsigned long long ga =
            (unsigned long long)(h + ((size_t)b * DM + d) * LL + l0 + lq * 4);
        unsigned lo = sbase + (unsigned)(d * LNP + lq * 4) * 4u;
        asm volatile("global_load_async_to_lds_b128 %0, %1, off"
                     :: "v"(lo), "v"(ga) : "memory");
    }
    asm volatile("s_wait_asynccnt 0" ::: "memory");
    __syncthreads();

    // ---- per-column LayerNorm (wave w -> columns 2w, 2w+1) ----
#pragma unroll
    for (int cc = 0; cc < 2; ++cc) {
        int col = 2 * w + cc;
        float v[16];
        float s1 = 0.0f, s2 = 0.0f;
#pragma unroll
        for (int j = 0; j < 16; ++j) {
            float x = sT[(lane + 32 * j) * LNP + col];
            v[j] = x; s1 += x; s2 += x * x;
        }
#pragma unroll
        for (int m = 1; m < 32; m <<= 1) {
            s1 += __shfl_xor(s1, m, 32);
            s2 += __shfl_xor(s2, m, 32);
        }
        float mu  = s1 * (1.0f / DM);
        float var = s2 * (1.0f / DM) - mu * mu;
        float rs  = rsqrtf(var + EPSLN);
#pragma unroll
        for (int j = 0; j < 16; ++j) {
            int d = lane + 32 * j;
            sT[d * LNP + col] = (v[j] - mu) * rs * g[layer * DM + d] + bta[layer * DM + d];
        }
    }
    __syncthreads();

    // ---- write back tile LDS -> memory (async from LDS) ----
#pragma unroll
    for (int j = 0; j < 8; ++j) {
        int e  = threadIdx.x + 256 * j;
        int d  = e >> 2;
        int lq = e & 3;
        unsigned long long ga =
            (unsigned long long)(h + ((size_t)b * DM + d) * LL + l0 + lq * 4);
        unsigned lo = sbase + (unsigned)(d * LNP + lq * 4) * 4u;
        asm volatile("global_store_async_from_lds_b128 %0, %1, off"
                     :: "v"(ga), "v"(lo) : "memory");
    }
    asm volatile("s_wait_asynccnt 0" ::: "memory");
}

// ---------------------------------------------------------------------------
// Kernel 5: mean over L -> hmH f16 (rows 0..7 of a 16x512 padded matrix)
// ---------------------------------------------------------------------------
__global__ void k_meanL(const float* __restrict__ h, _Float16* __restrict__ hmH) {
    int gw   = (blockIdx.x * blockDim.x + threadIdx.x) >> 5;
    int lane = threadIdx.x & 31;
    if (gw >= BB * DM) return;
    const float* p = h + (size_t)gw * LL;
    float s = 0.0f;
    for (int j = lane; j < LL; j += 32) s += p[j];
#pragma unroll
    for (int m = 1; m < 32; m <<= 1) s += __shfl_xor(s, m, 32);
    if (lane == 0) {
        int b = gw >> 9, d = gw & (DM - 1);
        hmH[b * DM + d] = (_Float16)(s * (1.0f / LL));
    }
}

// ---------------------------------------------------------------------------
// Kernel 6: stats via WMMA: (16x512 padded hm) x WstatsT (512x512) + bstats.
// ---------------------------------------------------------------------------
__global__ void k_stats_wmma(const _Float16* __restrict__ hmH,
                             const _Float16* __restrict__ WstatsH,
                             const float* __restrict__ bs,
                             float* __restrict__ out) {
    const int w    = threadIdx.x >> 5;
    const int lane = threadIdx.x & 31;
    const int o0   = w * 64;
    const int kbase = (lane < 16) ? 0 : 16;
    const int koffB = (lane < 16) ? 0 : 16;
    const _Float16* Arow  = hmH + (size_t)(lane & 15) * DM;
    const _Float16* Bbase = WstatsH + (size_t)(o0 + (lane & 15)) * DM + koffB;

    v8f acc0 = {}, acc1 = {}, acc2 = {}, acc3 = {};
#pragma unroll 2
    for (int kk = 0; kk < DM; kk += 32) {
        AFrag a;
        a.q[0] = *(const uint4*)(Arow + kk + kbase);
        a.q[1] = *(const uint4*)(Arow + kk + 8 + kbase);
#pragma unroll
        for (int t = 0; t < 4; ++t) {
            const _Float16* bp = Bbase + (size_t)(16 * t) * DM + kk;
            AFrag bf;
            bf.q[0] = *(const uint4*)(bp);
            bf.q[1] = *(const uint4*)(bp + 8);
            v8f* accp = (t == 0) ? &acc0 : (t == 1) ? &acc1 : (t == 2) ? &acc2 : &acc3;
            *accp = __builtin_amdgcn_wmma_f32_16x16x32_f16(
                false, a.v, false, bf.v, (short)0, *accp, false, false);
        }
    }
    if (lane < 16) {
#pragma unroll
        for (int t = 0; t < 4; ++t) {
            v8f acc = (t == 0) ? acc0 : (t == 1) ? acc1 : (t == 2) ? acc2 : acc3;
            int o = o0 + 16 * t + lane;
#pragma unroll
            for (int r = 0; r < 8; ++r) {              // r = batch index
                float s = acc[r] + bs[o];
                if (o < LATENT) out[r * LATENT + o] = s;
                else            out[BB * LATENT + r * LATENT + (o - LATENT)] = s;
            }
        }
    }
}

// ---------------------------------------------------------------------------
extern "C" void kernel_launch(void* const* d_in, const int* in_sizes, int n_in,
                              void* d_out, int out_size, void* d_ws, size_t ws_size,
                              hipStream_t stream) {
    (void)in_sizes; (void)n_in; (void)out_size; (void)ws_size;
    const float* x          = (const float*)d_in[0];
    const float* Wproj      = (const float*)d_in[1];
    const float* bproj      = (const float*)d_in[2];
    const float* pos_emb    = (const float*)d_in[3];
    const float* log_dt     = (const float*)d_in[4];
    const float* log_A_real = (const float*)d_in[5];
    const float* A_imag     = (const float*)d_in[6];
    const float* C_re       = (const float*)d_in[7];
    const float* C_im       = (const float*)d_in[8];
    const float* Dskip      = (const float*)d_in[9];
    const float* Wout       = (const float*)d_in[10];
    const float* bout       = (const float*)d_in[11];
    const float* ln_g       = (const float*)d_in[12];
    const float* ln_b       = (const float*)d_in[13];
    const float* Wstats     = (const float*)d_in[14];
    const float* bstats     = (const float*)d_in[15];
    float* out = (float*)d_out;

    // workspace carve-up (256B aligned slices)
    char* ws = (char*)d_ws;
    float*    h       = (float*)ws;                               // 64 MB  (B,D,L) f32
    _Float16* yactT   = (_Float16*)(ws + (size_t)67108864);       // 32 MB  (B,L,D) f16
    _Float16* WoutH   = (_Float16*)(ws + (size_t)100663296);      //  4 MB
    _Float16* WstatsH = (_Float16*)(ws + (size_t)104857600);      // 512 KB
    _Float16* hmH     = (_Float16*)(ws + (size_t)105381888);      // 16 KB (16x512 f16)

    // 0) weight conversions + pad
    k_cvt<<<(NLAY * 2 * DM * DM + 255) / 256, 256, 0, stream>>>(Wout, WoutH, NLAY * 2 * DM * DM);
    k_cvt<<<(2 * LATENT * DM + 255) / 256, 256, 0, stream>>>(Wstats, WstatsH, 2 * LATENT * DM);
    k_padzero<<<(8 * DM + 255) / 256, 256, 0, stream>>>(hmH);

    // 1) input projection + pos_emb
    k_proj<<<(BB * DM * LL) / 256, 256, 0, stream>>>(x, Wproj, bproj, pos_emb, h);

    // 2) layers
    for (int i = 0; i < NLAY; ++i) {
        k_ssm_gelu<<<(BB * DM) / 8, 256, 0, stream>>>(
            h, log_dt, log_A_real, A_imag, C_re, C_im, Dskip, yactT, i);
        dim3 g(LL / 64, DM / 64, BB);
        k_gemm_glu<<<g, 256, 0, stream>>>(WoutH, yactT, bout, h, i);
        k_layernorm<<<BB * (LL / 16), 256, 0, stream>>>(h, ln_g, ln_b, i);
    }

    // 3) mean over L + WMMA stats head
    k_meanL<<<(BB * DM * 32) / 256, 256, 0, stream>>>(h, hmH);
    k_stats_wmma<<<1, 256, 0, stream>>>(hmH, WstatsH, bstats, out);
}
